// InstructionGuidedPlasticityModule_5274219839741
// MI455X (gfx1250) — compile-verified, matrix operand order checked by reading, not run on hardware
//
#include <hip/hip_runtime.h>
#include <hip/hip_bf16.h>

typedef __attribute__((ext_vector_type(16))) _Float16 v16h;
typedef __attribute__((ext_vector_type(8)))  _Float16 v8h;
typedef __attribute__((ext_vector_type(8)))  float    v8f;

#if __has_builtin(__builtin_amdgcn_tensor_load_to_lds)
#define HAVE_TDM 1
typedef unsigned int tdm_u32x4 __attribute__((ext_vector_type(4)));
typedef int          tdm_i32x8 __attribute__((ext_vector_type(8)));
typedef int          tdm_i32x4 __attribute__((ext_vector_type(4)));
#else
#define HAVE_TDM 0
#endif

// Problem constants
constexpr int B_  = 4096;
constexpr int T_  = 32;
constexpr int D_  = 512;
constexpr int H_  = 256;   // LSTM hidden per direction
constexpr int I_  = 512;   // instr width
constexpr int S_  = 16;    // slots
constexpr int G4_ = 1024;  // 4*H
constexpr int C_  = 1024;  // D + I
constexpr int DH_ = 256;   // D/2

// ---------------------------------------------------------------------------
// One K=32 step: 2 A-frags x 4 B-frags -> 8 v_wmma_f32_16x16x32_f16
// ---------------------------------------------------------------------------
__device__ __forceinline__ void wmma_step(
    v8h a00, v8h a01, v8h a10, v8h a11,
    const _Float16* __restrict__ w0, const _Float16* __restrict__ w1,
    const _Float16* __restrict__ w2, const _Float16* __restrict__ w3,
    int o0, int o1, v8f (&acc)[2][4])
{
    v16h a[2];
#pragma unroll
    for (int i = 0; i < 8; ++i) {
        a[0][i] = a00[i]; a[0][i + 8] = a01[i];
        a[1][i] = a10[i]; a[1][i + 8] = a11[i];
    }
    const _Float16* ws[4] = {w0, w1, w2, w3};
#pragma unroll
    for (int nt = 0; nt < 4; ++nt) {
        v8h b0 = *(const v8h*)(ws[nt] + o0);
        v8h b1 = *(const v8h*)(ws[nt] + o1);
        v16h b;
#pragma unroll
        for (int i = 0; i < 8; ++i) { b[i] = b0[i]; b[i + 8] = b1[i]; }
#pragma unroll
        for (int mt = 0; mt < 2; ++mt)
            acc[mt][nt] = __builtin_amdgcn_wmma_f32_16x16x32_f16(
                false, a[mt], false, b, (short)0, acc[mt][nt], false, false);
    }
}

// ---------------------------------------------------------------------------
// Register-blocked WMMA GEMM:  out[M,N] = A[M,K] (f16 rm) @ W[N,K]^T (f16)
//  + optional f32 bias0/bias1, optional f16 C-init, relu, f32/f16 outputs.
// Block = 256 thr = 8 waves as 2(M) x 4(N); wave tile 32x64 -> block 64x256.
// useLds: A panel (64 x K f16, contiguous) staged into LDS via the Tensor
// Data Mover (tensor_load_to_lds + s_wait_tensorcnt), then read as ds_load;
// reused by the 4 N-wave columns (4x less global A traffic).
// ---------------------------------------------------------------------------
__global__ __launch_bounds__(256)
void gemm_wmma_kernel(const _Float16* __restrict__ A, long sAb,
                      const _Float16* __restrict__ W, long sWb,
                      const float* __restrict__ bias0,
                      const float* __restrict__ bias1, long sBiasb,
                      const _Float16* __restrict__ Cinit, int ldCinit,
                      float* __restrict__ outF, _Float16* __restrict__ outH,
                      long sOutb, int M, int N, int K, int relu, int useLds)
{
    extern __shared__ _Float16 ldsA[];   // 64 x K panel when useLds

    const int zb = blockIdx.z;
    A += zb * sAb;
    W += zb * sWb;
    if (bias0) bias0 += zb * sBiasb;
    if (bias1) bias1 += zb * sBiasb;
    if (outF)  outF  += zb * sOutb;
    if (outH)  outH  += zb * sOutb;

    const int lane  = threadIdx.x & 31;
    const int wave  = threadIdx.x >> 5;
    const int waveM = wave >> 2;                  // 0..1
    const int waveN = wave & 3;                   // 0..3
    const int mWave = blockIdx.y * 64 + waveM * 32;
    const int nWave = blockIdx.x * 256 + waveN * 64;
    const int lrow  = lane & 15;
    const int koff  = (lane >> 4) * 8;            // ISA 16-bit A-layout K split

    if (useLds) {
#if HAVE_TDM
        if (threadIdx.x < 32) {   // wave 0 issues one TDM 2D tile load
            unsigned long long ga =
                (unsigned long long)(const void*)(A + (long)blockIdx.y * 64 * K);
            tdm_u32x4 g0;
            g0[0] = 1u;                                   // count=1 (valid D#)
            g0[1] = 0u;                                   // lds_addr = dyn-LDS base
            g0[2] = (unsigned)(ga & 0xFFFFFFFFull);       // global_addr lo
            g0[3] = (unsigned)((ga >> 32) & 0x01FFFFFFull) | (2u << 30); // hi|type=2
            tdm_i32x8 g1;
            g1[0] = 0x00010000;                           // data_size = 2 bytes
            g1[1] = (K & 0xFFFF) << 16;                   // tensor_dim0 = K
            g1[2] = (64 << 16);                           // tensor_dim1 = 64
            g1[3] = (K & 0xFFFF) << 16;                   // tile_dim0 = K
            g1[4] = 64;                                   // tile_dim1 = 64
            g1[5] = K;                                    // tensor_dim0_stride = K
            g1[6] = 0; g1[7] = 0;
            tdm_i32x4 gz; gz[0] = 0; gz[1] = 0; gz[2] = 0; gz[3] = 0;
            tdm_i32x8 gz8;
#pragma unroll
            for (int i = 0; i < 8; ++i) gz8[i] = 0;
            __builtin_amdgcn_tensor_load_to_lds(g0, g1, gz, gz, gz8, 0);
            __builtin_amdgcn_s_wait_tensorcnt(0);
        }
        __syncthreads();
#else
        {   // cooperative 128b copy fallback (panel is contiguous 64*K halves)
            const _Float16* src = A + (long)blockIdx.y * 64 * K;
            const int total = (64 * K) >> 3;
            for (int i = threadIdx.x; i < total; i += 256)
                ((v8h*)ldsA)[i] = ((const v8h*)src)[i];
            __syncthreads();
        }
#endif
    }

    const _Float16* wR0 = W + (long)(nWave +  0 + lrow) * K;
    const _Float16* wR1 = W + (long)(nWave + 16 + lrow) * K;
    const _Float16* wR2 = W + (long)(nWave + 32 + lrow) * K;
    const _Float16* wR3 = W + (long)(nWave + 48 + lrow) * K;

    v8f acc[2][4] = {};

    if (useLds) {
        const int l0 = (waveM * 32 + lrow) * K;
        const int l1 = l0 + 16 * K;
        for (int k0 = 0; k0 < K; k0 += 32) {
            const int o0 = k0 + koff, o1 = k0 + 16 + koff;
            v8h a00 = *(const v8h*)(&ldsA[l0 + o0]);
            v8h a01 = *(const v8h*)(&ldsA[l0 + o1]);
            v8h a10 = *(const v8h*)(&ldsA[l1 + o0]);
            v8h a11 = *(const v8h*)(&ldsA[l1 + o1]);
            if (k0 + 32 < K) {
                __builtin_prefetch(wR0 + k0 + 32, 0, 3);   // global_prefetch_b8
                __builtin_prefetch(wR2 + k0 + 32, 0, 3);
            }
            wmma_step(a00, a01, a10, a11, wR0, wR1, wR2, wR3, o0, o1, acc);
        }
    } else {
        const _Float16* aR0 = A + (long)(mWave + lrow) * K;
        const _Float16* aR1 = A + (long)(mWave + 16 + lrow) * K;
        for (int k0 = 0; k0 < K; k0 += 32) {
            const int o0 = k0 + koff, o1 = k0 + 16 + koff;
            v8h a00 = *(const v8h*)(aR0 + o0);
            v8h a01 = *(const v8h*)(aR0 + o1);
            v8h a10 = *(const v8h*)(aR1 + o0);
            v8h a11 = *(const v8h*)(aR1 + o1);
            if (k0 + 32 < K) {
                __builtin_prefetch(aR0 + k0 + 32, 0, 3);
                __builtin_prefetch(wR0 + k0 + 32, 0, 3);
                __builtin_prefetch(wR2 + k0 + 32, 0, 3);
            }
            wmma_step(a00, a01, a10, a11, wR0, wR1, wR2, wR3, o0, o1, acc);
        }
    }

    // C/D layout: VGPR r, lanes 0-15 -> M=base+r, lanes 16-31 -> M=base+8+r
#pragma unroll
    for (int mt = 0; mt < 2; ++mt) {
        const int rBase = mWave + mt * 16 + ((lane >> 4) * 8);
#pragma unroll
        for (int nt = 0; nt < 4; ++nt) {
            const int n = nWave + nt * 16 + lrow;
            float bv = 0.f;
            if (bias0) bv += bias0[n];
            if (bias1) bv += bias1[n];
#pragma unroll
            for (int r = 0; r < 8; ++r) {
                const int row = rBase + r;
                float v = acc[mt][nt][r] + bv;
                if (Cinit) v += (float)Cinit[(long)row * ldCinit + n];
                if (relu)  v = fmaxf(v, 0.f);
                const long oi = (long)row * N + n;
                if (outF) outF[oi] = v;
                if (outH) outH[oi] = (_Float16)v;
            }
        }
    }
}

// ---------------------------------------------------------------------------
// Elementwise / helper kernels
// ---------------------------------------------------------------------------
__global__ void zero_f32_kernel(float* p, long n) {
    long i = (long)blockIdx.x * blockDim.x + threadIdx.x;
    if (i < n) p[i] = 0.f;
}
__global__ void zero_f16_kernel(_Float16* p, long n) {
    long i = (long)blockIdx.x * blockDim.x + threadIdx.x;
    if (i < n) p[i] = (_Float16)0.f;
}
__global__ void f32_to_f16_kernel(const float* __restrict__ in,
                                  _Float16* __restrict__ out, long n) {
    long i = (long)blockIdx.x * blockDim.x + threadIdx.x;
    if (i < n) out[i] = (_Float16)in[i];
}
// in: [S][K][N] f32  ->  out: [S][N][K] f16
__global__ void transpose_f16_kernel(const float* __restrict__ in,
                                     _Float16* __restrict__ out,
                                     int K, int N, long total) {
    long idx = (long)blockIdx.x * blockDim.x + threadIdx.x;
    if (idx >= total) return;
    long sn = (long)K * N;
    int s = (int)(idx / sn);
    long r = idx - (long)s * sn;
    int k = (int)(r / N);
    int nn = (int)(r - (long)k * N);
    out[(long)s * sn + (long)nn * K + k] = (_Float16)in[idx];
}
// Embedding gather + f16 convert
__global__ void embed_kernel(const int* __restrict__ tok,
                             const float* __restrict__ embed,
                             _Float16* __restrict__ out) {
    long idx = (long)blockIdx.x * blockDim.x + threadIdx.x;
    if (idx >= (long)B_ * T_ * D_) return;
    long bt = idx / D_;
    int d = (int)(idx - bt * D_);
    int t = tok[bt];
    out[idx] = (_Float16)embed[(long)t * D_ + d];
}
// LSTM cell: gates [B,4H] f32 -> c update, h -> f16 for next GEMM, hsum accum
__global__ void lstm_cell_kernel(const float* __restrict__ gates,
                                 float* __restrict__ c,
                                 _Float16* __restrict__ hH,
                                 float* __restrict__ hsum, int dirOff) {
    int idx = blockIdx.x * blockDim.x + threadIdx.x;   // B*H
    if (idx >= B_ * H_) return;
    int b = idx >> 8;
    int j = idx & (H_ - 1);
    const float* g = gates + (long)b * G4_;
    float ig = g[j], fg = g[H_ + j], gg = g[2 * H_ + j], og = g[3 * H_ + j];
    float si = 1.f / (1.f + __expf(-ig));
    float sf = 1.f / (1.f + __expf(-fg));
    float so = 1.f / (1.f + __expf(-og));
    float tg = tanhf(gg);
    float cv = sf * c[idx] + si * tg;
    float hv = so * tanhf(cv);
    c[idx] = cv;
    hH[idx] = (_Float16)hv;
    hsum[(long)b * I_ + dirOff + j] += hv;
}
// instr = hsum/T; build comb=[x,instr] f16, xH f16, instrH f16, instrF f32
__global__ void instr_comb_kernel(const float* __restrict__ x,
                                  const float* __restrict__ hsum,
                                  float* __restrict__ instrF,
                                  _Float16* __restrict__ instrH,
                                  _Float16* __restrict__ xH,
                                  _Float16* __restrict__ comb) {
    int idx = blockIdx.x * blockDim.x + threadIdx.x;   // B*1024
    if (idx >= B_ * C_) return;
    int b = idx >> 10;
    int j = idx & (C_ - 1);
    if (j < D_) {
        float v = x[(long)b * D_ + j];
        xH[(long)b * D_ + j] = (_Float16)v;
        comb[idx] = (_Float16)v;
    } else {
        int i = j - D_;
        float v = hsum[(long)b * I_ + i] * (1.f / T_);
        instrF[(long)b * I_ + i] = v;
        instrH[(long)b * I_ + i] = (_Float16)v;
        comb[idx] = (_Float16)v;
    }
}
// Two-stage deterministic reduction for mean / norm of instr
__global__ void reduce_part_kernel(const float* __restrict__ v, long n,
                                   float* __restrict__ partials) {
    __shared__ float s0[256], s1[256];
    float a = 0.f, q = 0.f;
    for (long i = (long)blockIdx.x * blockDim.x + threadIdx.x; i < n;
         i += (long)gridDim.x * blockDim.x) {
        float t = v[i]; a += t; q += t * t;
    }
    s0[threadIdx.x] = a; s1[threadIdx.x] = q;
    __syncthreads();
    for (int o = 128; o > 0; o >>= 1) {
        if ((int)threadIdx.x < o) {
            s0[threadIdx.x] += s0[threadIdx.x + o];
            s1[threadIdx.x] += s1[threadIdx.x + o];
        }
        __syncthreads();
    }
    if (threadIdx.x == 0) {
        partials[blockIdx.x * 2]     = s0[0];
        partials[blockIdx.x * 2 + 1] = s1[0];
    }
}
__global__ void scalar_kernel(const float* __restrict__ partials, int nblk,
                              float* __restrict__ scal) {
    __shared__ float s0[256], s1[256];
    float a = 0.f, q = 0.f;
    for (int i = threadIdx.x; i < nblk; i += 256) {
        a += partials[i * 2]; q += partials[i * 2 + 1];
    }
    s0[threadIdx.x] = a; s1[threadIdx.x] = q;
    __syncthreads();
    for (int o = 128; o > 0; o >>= 1) {
        if ((int)threadIdx.x < o) {
            s0[threadIdx.x] += s0[threadIdx.x + o];
            s1[threadIdx.x] += s1[threadIdx.x + o];
        }
        __syncthreads();
    }
    if (threadIdx.x == 0) {
        float m = s0[0] / (float)((long)B_ * I_);
        float nn = sqrtf(s1[0]);
        float cm = m > 0.1f ? 1.5f
                 : (m < -0.1f ? 0.8f : (nn > 1.0f ? 1.2f : 1.0f));
        scal[0] = 2.f * cm * cm * 0.675f;   // ps
    }
}
// plast[b,s] = sigmoid( dot(hp[s][b][:DH], Wp2[s]) + bp2[s] ) * 0.1
__global__ void plast_kernel(const float* __restrict__ hp,
                             const float* __restrict__ Wp2,
                             const float* __restrict__ bp2,
                             float* __restrict__ plast) {
    int w = (blockIdx.x * blockDim.x + threadIdx.x) >> 5;
    int lane = threadIdx.x & 31;
    if (w >= B_ * S_) return;
    int b = w / S_, s = w - b * S_;
    const float* hrow = hp + ((long)s * B_ + b) * DH_;
    const float* wrow = Wp2 + (long)s * DH_;
    float sum = 0.f;
    for (int j = lane; j < DH_; j += 32) sum += hrow[j] * wrow[j];
    for (int o = 16; o; o >>= 1) sum += __shfl_xor(sum, o, 32);
    if (lane == 0)
        plast[b * S_ + s] = 0.1f / (1.f + __expf(-(sum + bp2[s])));
}
// logits[b,s] = dot(instr[b], Wsel[:,s]) + bsel[s]
__global__ void logits_kernel(const float* __restrict__ instrF,
                              const float* __restrict__ Wsel,
                              const float* __restrict__ bsel,
                              float* __restrict__ logits) {
    int w = (blockIdx.x * blockDim.x + threadIdx.x) >> 5;
    int lane = threadIdx.x & 31;
    if (w >= B_ * S_) return;
    int b = w / S_, s = w - b * S_;
    const float* irow = instrF + (long)b * I_;
    float sum = 0.f;
    for (int j = lane; j < I_; j += 32) sum += irow[j] * Wsel[(long)j * S_ + s];
    for (int o = 16; o; o >>= 1) sum += __shfl_xor(sum, o, 32);
    if (lane == 0) logits[b * S_ + s] = sum + bsel[s];
}
__global__ void softmax_kernel(const float* __restrict__ logits,
                               float* __restrict__ attn) {
    int b = blockIdx.x * blockDim.x + threadIdx.x;
    if (b >= B_) return;
    const float* l = logits + (long)b * S_;
    float mx = l[0];
    for (int s = 1; s < S_; ++s) mx = fmaxf(mx, l[s]);
    float e[S_], sum = 0.f;
    for (int s = 0; s < S_; ++s) { e[s] = __expf(l[s] - mx); sum += e[s]; }
    float inv = 1.f / sum;
    float* a = attn + (long)b * S_;
    for (int s = 0; s < S_; ++s) a[s] = e[s] * inv;
}
// LayerNorm stats of gateRaw rows -> mean, rstd
__global__ void lnstats_kernel(const float* __restrict__ gr,
                               float* __restrict__ stats) {
    int w = (blockIdx.x * blockDim.x + threadIdx.x) >> 5;
    int lane = threadIdx.x & 31;
    if (w >= B_ * S_) return;
    int b = w / S_, s = w - b * S_;
    const float* row = gr + ((long)s * B_ + b) * D_;
    float a = 0.f, q = 0.f;
    for (int d = lane; d < D_; d += 32) { float v = row[d]; a += v; q += v * v; }
    for (int o = 16; o; o >>= 1) { a += __shfl_xor(a, o, 32); q += __shfl_xor(q, o, 32); }
    if (lane == 0) {
        float m = a * (1.f / D_);
        float var = q * (1.f / D_) - m * m;
        stats[(b * S_ + s) * 2]     = m;
        stats[(b * S_ + s) * 2 + 1] = rsqrtf(var + 1e-5f);
    }
}
// out[b,d] = sum_s attn * (0.5*x + plast * transformed * tanh(LN(gateRaw)) * ps)
__global__ void combine_kernel(const float* __restrict__ x,
                               const float* __restrict__ transformed,
                               const float* __restrict__ gateRaw,
                               const float* __restrict__ stats,
                               const float* __restrict__ lng,
                               const float* __restrict__ lnb,
                               const float* __restrict__ plast,
                               const float* __restrict__ attn,
                               const float* __restrict__ scal,
                               float* __restrict__ out) {
    int idx = blockIdx.x * blockDim.x + threadIdx.x;   // B*D
    if (idx >= B_ * D_) return;
    int b = idx >> 9;
    int d = idx & (D_ - 1);
    float ps = scal[0];
    float xv = x[idx];
    float acc = 0.f;
#pragma unroll 4
    for (int s = 0; s < S_; ++s) {
        long o = ((long)s * B_ + b) * D_ + d;
        float mean = stats[(b * S_ + s) * 2];
        float rstd = stats[(b * S_ + s) * 2 + 1];
        float g = tanhf((gateRaw[o] - mean) * rstd * lng[s * D_ + d] + lnb[s * D_ + d]);
        float so = xv * 0.5f + plast[b * S_ + s] * (transformed[o] * g) * ps;
        acc += attn[b * S_ + s] * so;
    }
    out[idx] = acc;
}

// ---------------------------------------------------------------------------
// Host orchestration
// ---------------------------------------------------------------------------
static inline size_t alignUp256(size_t v) { return (v + 255) & ~(size_t)255; }
static inline int nblk(long n, int t) { return (int)((n + t - 1) / t); }

static void launch_gemm(hipStream_t st,
                        const _Float16* A, long sA,
                        const _Float16* W, long sW,
                        const float* b0, const float* b1, long sB,
                        const _Float16* Ci, int ldCi,
                        float* oF, _Float16* oH, long sO,
                        int M, int N, int K, int relu, int batch) {
    const int useLds = (K <= 512) ? 1 : 0;
    const size_t sh = useLds ? (size_t)64 * K * sizeof(_Float16) : 0;
    dim3 g(N / 256, M / 64, batch), blk(256);
    gemm_wmma_kernel<<<g, blk, sh, st>>>(A, sA, W, sW, b0, b1, sB,
                                         Ci, ldCi, oF, oH, sO,
                                         M, N, K, relu, useLds);
}

extern "C" void kernel_launch(void* const* d_in, const int* in_sizes, int n_in,
                              void* d_out, int out_size, void* d_ws, size_t ws_size,
                              hipStream_t stream) {
    (void)in_sizes; (void)n_in; (void)out_size; (void)ws_size;

    const int*   tok    = (const int*)  d_in[0];
    const float* x      = (const float*)d_in[1];
    const float* embed  = (const float*)d_in[2];
    const float* Wih_f  = (const float*)d_in[3];
    const float* Whh_f  = (const float*)d_in[4];
    const float* bih_f  = (const float*)d_in[5];
    const float* bhh_f  = (const float*)d_in[6];
    const float* Wih_b  = (const float*)d_in[7];
    const float* Whh_b  = (const float*)d_in[8];
    const float* bih_b  = (const float*)d_in[9];
    const float* bhh_b  = (const float*)d_in[10];
    const float* fastW  = (const float*)d_in[11];
    const float* fastb  = (const float*)d_in[12];
    const float* Wg     = (const float*)d_in[13];
    const float* bg     = (const float*)d_in[14];
    const float* lng    = (const float*)d_in[15];
    const float* lnb    = (const float*)d_in[16];
    const float* Wp1    = (const float*)d_in[17];
    const float* bp1    = (const float*)d_in[18];
    const float* Wp2    = (const float*)d_in[19];
    const float* bp2    = (const float*)d_in[20];
    const float* Wsel   = (const float*)d_in[21];
    const float* bsel   = (const float*)d_in[22];
    float* out = (float*)d_out;

    // ---- workspace bump allocator ----
    char* wp = (char*)d_ws;
    auto alloc = [&](size_t bytes) -> char* {
        char* p = wp; wp += alignUp256(bytes); return p;
    };
    _Float16* wWihF  = (_Float16*)alloc((size_t)G4_ * D_ * 2);
    _Float16* wWhhF  = (_Float16*)alloc((size_t)G4_ * H_ * 2);
    _Float16* wWihB  = (_Float16*)alloc((size_t)G4_ * D_ * 2);
    _Float16* wWhhB  = (_Float16*)alloc((size_t)G4_ * H_ * 2);
    _Float16* wWp1t  = (_Float16*)alloc((size_t)S_ * DH_ * C_ * 2);   // [S][DH][C]
    _Float16* wFastWt= (_Float16*)alloc((size_t)S_ * D_ * D_ * 2);    // [S][D][D]
    _Float16* wWgt   = (_Float16*)alloc((size_t)S_ * D_ * I_ * 2);    // [S][D][I]
    _Float16* embH   = (_Float16*)alloc((size_t)B_ * T_ * D_ * 2);
    _Float16* gxF    = (_Float16*)alloc((size_t)B_ * T_ * G4_ * 2);
    _Float16* gxB    = (_Float16*)alloc((size_t)B_ * T_ * G4_ * 2);
    float*    gates  = (float*)   alloc((size_t)B_ * G4_ * 4);
    float*    cState = (float*)   alloc((size_t)B_ * H_ * 4);
    _Float16* hH     = (_Float16*)alloc((size_t)B_ * H_ * 2);
    float*    hsum   = (float*)   alloc((size_t)B_ * I_ * 4);
    float*    instrF = (float*)   alloc((size_t)B_ * I_ * 4);
    _Float16* instrH = (_Float16*)alloc((size_t)B_ * I_ * 2);
    _Float16* xH     = (_Float16*)alloc((size_t)B_ * D_ * 2);
    _Float16* comb   = (_Float16*)alloc((size_t)B_ * C_ * 2);
    float*    hp     = (float*)   alloc((size_t)S_ * B_ * DH_ * 4);
    float*    transf = (float*)   alloc((size_t)S_ * B_ * D_ * 4);
    float*    gateRw = (float*)   alloc((size_t)S_ * B_ * D_ * 4);
    float*    plast  = (float*)   alloc((size_t)B_ * S_ * 4);
    float*    logits = (float*)   alloc((size_t)B_ * S_ * 4);
    float*    attn   = (float*)   alloc((size_t)B_ * S_ * 4);
    float*    stats  = (float*)   alloc((size_t)B_ * S_ * 2 * 4);
    float*    parts  = (float*)   alloc((size_t)1024 * 2 * 4);
    float*    scal   = (float*)   alloc((size_t)4 * 4);

    const int TPB = 256;

    // ---- 1. weight conversion (f32 -> f16, W already [N,K] for X@W^T) ----
    f32_to_f16_kernel<<<nblk((long)G4_ * D_, TPB), TPB, 0, stream>>>(Wih_f, wWihF, (long)G4_ * D_);
    f32_to_f16_kernel<<<nblk((long)G4_ * H_, TPB), TPB, 0, stream>>>(Whh_f, wWhhF, (long)G4_ * H_);
    f32_to_f16_kernel<<<nblk((long)G4_ * D_, TPB), TPB, 0, stream>>>(Wih_b, wWihB, (long)G4_ * D_);
    f32_to_f16_kernel<<<nblk((long)G4_ * H_, TPB), TPB, 0, stream>>>(Whh_b, wWhhB, (long)G4_ * H_);
    // transpose [S][K][N] -> [S][N][K]
    transpose_f16_kernel<<<nblk((long)S_ * C_ * DH_, TPB), TPB, 0, stream>>>(Wp1,   wWp1t,   C_, DH_, (long)S_ * C_ * DH_);
    transpose_f16_kernel<<<nblk((long)S_ * D_ * D_,  TPB), TPB, 0, stream>>>(fastW, wFastWt, D_, D_,  (long)S_ * D_ * D_);
    transpose_f16_kernel<<<nblk((long)S_ * I_ * D_,  TPB), TPB, 0, stream>>>(Wg,    wWgt,    I_, D_,  (long)S_ * I_ * D_);

    // ---- 2. embedding gather ----
    embed_kernel<<<nblk((long)B_ * T_ * D_, TPB), TPB, 0, stream>>>(tok, embed, embH);

    // ---- 3. LSTM input projections (bias folded in): [B*T,D] @ Wih^T ----
    launch_gemm(stream, embH, 0, wWihF, 0, bih_f, bhh_f, 0,
                nullptr, 0, nullptr, gxF, 0, B_ * T_, G4_, D_, 0, 1);
    launch_gemm(stream, embH, 0, wWihB, 0, bih_b, bhh_b, 0,
                nullptr, 0, nullptr, gxB, 0, B_ * T_, G4_, D_, 0, 1);

    // ---- 4. LSTM recurrence ----
    zero_f32_kernel<<<nblk((long)B_ * I_, TPB), TPB, 0, stream>>>(hsum, (long)B_ * I_);
    for (int dir = 0; dir < 2; ++dir) {
        const _Float16* gx  = dir == 0 ? gxF : gxB;
        const _Float16* Whh = dir == 0 ? wWhhF : wWhhB;
        const int dirOff = dir == 0 ? 0 : H_;
        zero_f32_kernel<<<nblk((long)B_ * H_, TPB), TPB, 0, stream>>>(cState, (long)B_ * H_);
        zero_f16_kernel<<<nblk((long)B_ * H_, TPB), TPB, 0, stream>>>(hH, (long)B_ * H_);
        for (int t = 0; t < T_; ++t) {
            const int tIdx = dir == 0 ? t : (T_ - 1 - t);
            // gates = hH @ Whh^T + gx[:, tIdx, :]
            launch_gemm(stream, hH, 0, Whh, 0, nullptr, nullptr, 0,
                        gx + (long)tIdx * G4_, T_ * G4_,
                        gates, nullptr, 0, B_, G4_, H_, 0, 1);
            lstm_cell_kernel<<<nblk((long)B_ * H_, TPB), TPB, 0, stream>>>(
                gates, cState, hH, hsum, dirOff);
        }
    }

    // ---- 5. instr, comb, scalar modulation ----
    instr_comb_kernel<<<nblk((long)B_ * C_, TPB), TPB, 0, stream>>>(
        x, hsum, instrF, instrH, xH, comb);
    reduce_part_kernel<<<1024, TPB, 0, stream>>>(instrF, (long)B_ * I_, parts);
    scalar_kernel<<<1, TPB, 0, stream>>>(parts, 1024, scal);

    // ---- 6. per-slot GEMMs (batched over grid.z = S) ----
    // hp = relu(comb @ Wp1[s] + bp1[s])          [S][B][DH]
    launch_gemm(stream, comb, 0, wWp1t, (long)DH_ * C_, bp1, nullptr, DH_,
                nullptr, 0, hp, nullptr, (long)B_ * DH_, B_, DH_, C_, 1, S_);
    // transformed = x @ fastW[s] + fastb[s]        [S][B][D]
    launch_gemm(stream, xH, 0, wFastWt, (long)D_ * D_, fastb, nullptr, D_,
                nullptr, 0, transf, nullptr, (long)B_ * D_, B_, D_, D_, 0, S_);
    // gateRaw = instr @ Wg[s] + bg[s]              [S][B][D]
    launch_gemm(stream, instrH, 0, wWgt, (long)D_ * I_, bg, nullptr, D_,
                nullptr, 0, gateRw, nullptr, (long)B_ * D_, B_, D_, I_, 0, S_);

    // ---- 7. plast, attn, LN stats, final combine ----
    plast_kernel<<<nblk((long)B_ * S_ * 32, TPB), TPB, 0, stream>>>(hp, Wp2, bp2, plast);
    logits_kernel<<<nblk((long)B_ * S_ * 32, TPB), TPB, 0, stream>>>(instrF, Wsel, bsel, logits);
    softmax_kernel<<<nblk((long)B_, TPB), TPB, 0, stream>>>(logits, attn);
    lnstats_kernel<<<nblk((long)B_ * S_ * 32, TPB), TPB, 0, stream>>>(gateRw, stats);
    combine_kernel<<<nblk((long)B_ * D_, TPB), TPB, 0, stream>>>(
        x, transf, gateRw, stats, lng, lnb, plast, attn, scal, out);
}
